// TJDLayer_3393024163846
// MI455X (gfx1250) — matrix operand order repaired and checked by reading, not compiled, not run on hardware
//
#include <hip/hip_runtime.h>
#include <hip/hip_bf16.h>

#define EMB    2048
#define RANK   4
#define VOCAB  8192
#define NH     4
#define BATCH  256
#define NCOLS  (VOCAB * RANK * RANK)   // 131072
#define EPSV   1e-10f

#define NT   64    // columns per workgroup (one i-block slice: 16 v's x 4 j's)
#define KT   32    // K step per WMMA
#define LROW 40    // padded LDS row stride (halfs) to spread banks

typedef __attribute__((ext_vector_type(16))) _Float16 v16h;
typedef __attribute__((ext_vector_type(8)))  _Float16 v8h;
typedef __attribute__((ext_vector_type(8)))  float    v8f;

// ---------------------------------------------------------------- x -> f16
__global__ void convert_x_kernel(const float* __restrict__ x, _Float16* __restrict__ xh) {
    int i = blockIdx.x * blockDim.x + threadIdx.x;      // 2048 blocks * 256
    if (i < BATCH * EMB) xh[i] = (_Float16)x[i];
}

// ---------------------------------------------------------------- zero cm
__global__ void zero_cm_kernel(float* __restrict__ cm) {
    int i = blockIdx.x * blockDim.x + threadIdx.x;      // 16 * 256 = 4096
    if (i < BATCH * RANK * RANK) cm[i] = 0.0f;
}

static __device__ __forceinline__ v16h load_frag16(const _Float16* p) {
    v8h lo = *(const v8h*)(p);
    v8h hi = *(const v8h*)(p + 8);
    return __builtin_shufflevector(lo, hi, 0,1,2,3,4,5,6,7,8,9,10,11,12,13,14,15);
}

// ------------------------------------------------- main fused WMMA GEMM + Σ_v relu
// grid = NCOLS/NT = 2048 WGs, block = 512 (16 waves). Wave w owns rows [16w,16w+16),
// all 64 columns of the WG (4 WMMA N-subtiles). w_vocab streamed from HBM once.
__global__ __launch_bounds__(512) void gemm_cm_kernel(const _Float16* __restrict__ xh,
                                                      const float* __restrict__ w,
                                                      float* __restrict__ cm) {
    __shared__ _Float16 lb[NT * LROW];                  // staged B slab, f16, [col][k]

    const int t       = threadIdx.x;
    const int wv      = t >> 5;                         // wave id 0..15
    const int L       = t & 31;                         // lane id
    const int colbase = blockIdx.x * NT;

    // staging assignment: col = t%64, k4 = 4*(t/64)  (coalesced 256B row segments)
    const int stage_col = t & 63;
    const int stage_k4  = (t >> 6) << 2;

    // A-fragment addressing (contiguous K order, matches B):
    // lanes 0-15: row M=L,   elements = K {0..15}
    // lanes16-31: row M=L-16,elements = K {16..31}
    const int  half   = (L < 16) ? 0 : 1;
    const int  mrow   = 16 * wv + (L & 15);
    const _Float16* arow = xh + (size_t)mrow * EMB + (half ? 16 : 0);

    // B-fragment LDS base: col = 16n + (L&15); lanes<16: K k0..k0+15, lanes>=16: +16
    const _Float16* bbase = lb + (L & 15) * LROW + (half ? 16 : 0);

    v8f acc0 = {}, acc1 = {}, acc2 = {}, acc3 = {};

    for (int k0 = 0; k0 < EMB; k0 += KT) {
        // ---- stage 32 x 64 f32 slab -> f16 LDS (each thread: 4 strided rows)
        const float* wp = w + (size_t)(k0 + stage_k4) * NCOLS + colbase + stage_col;
        float f0 = wp[0];
        float f1 = wp[NCOLS];
        float f2 = wp[2 * NCOLS];
        float f3 = wp[3 * NCOLS];
        // WGP-scope prefetch of the next slab: pulls into all cache levels so the
        // next iteration's staging loads hit the WGP cache, not just GL2.
        if (k0 + KT < EMB) __builtin_prefetch(wp + (size_t)KT * NCOLS, 0, 3);
        _Float16* lp = lb + stage_col * LROW + stage_k4;
        lp[0] = (_Float16)f0; lp[1] = (_Float16)f1;
        lp[2] = (_Float16)f2; lp[3] = (_Float16)f3;
        __syncthreads();

        // ---- A fragment: two 16B loads of this lane's row chunks
        v16h a = load_frag16(arow + k0);

        // ---- four B fragments + four WMMAs
        v16h b0 = load_frag16(bbase);
        v16h b1 = load_frag16(bbase + 16 * LROW);
        v16h b2 = load_frag16(bbase + 32 * LROW);
        v16h b3 = load_frag16(bbase + 48 * LROW);

        acc0 = __builtin_amdgcn_wmma_f32_16x16x32_f16(false, a, false, b0, (short)0, acc0, false, false);
        acc1 = __builtin_amdgcn_wmma_f32_16x16x32_f16(false, a, false, b1, (short)0, acc1, false, false);
        acc2 = __builtin_amdgcn_wmma_f32_16x16x32_f16(false, a, false, b2, (short)0, acc2, false, false);
        acc3 = __builtin_amdgcn_wmma_f32_16x16x32_f16(false, a, false, b3, (short)0, acc3, false, false);
        __syncthreads();
    }

    // ---- epilogue: relu, reduce over v (lanes differing in bits 2,3 share j), atomic cm add
    // C/D layout: VGPR r, lane L -> row M = r + (L<16?0:8), col N = L&15
    const int i = colbase >> 15;      // which RANK-i block (colbase / 32768)
    const int j = L & 3;
    float part[8];
    #pragma unroll
    for (int r = 0; r < 8; ++r) {
        float s = fmaxf(acc0[r], 0.f) + fmaxf(acc1[r], 0.f)
                + fmaxf(acc2[r], 0.f) + fmaxf(acc3[r], 0.f);
        s += __shfl_xor(s, 4, 32);
        s += __shfl_xor(s, 8, 32);
        part[r] = s;
    }
    if ((L & 12) == 0) {              // lanes 0-3 and 16-19 hold the j-sums
        const int mbase = 16 * wv + (half ? 8 : 0);
        #pragma unroll
        for (int r = 0; r < 8; ++r)
            atomicAdd(cm + (mbase + r) * (RANK * RANK) + i * RANK + j, part[r]);
    }
}

// ---------------------------------------------------------------- alpha/beta
__global__ void alphabeta_kernel(const float* __restrict__ x,
                                 const float* __restrict__ wa,
                                 const float* __restrict__ wb,
                                 float* __restrict__ alpha, float* __restrict__ beta) {
    __shared__ float red[256 * 8];
    const int b = blockIdx.x, t = threadIdx.x;
    float s[8] = {0,0,0,0,0,0,0,0};
    for (int k = t; k < EMB; k += 256) {
        float xv = x[(size_t)b * EMB + k];
        const float* a4 = wa + k * RANK;
        const float* b4 = wb + k * RANK;
        s[0] += xv * a4[0]; s[1] += xv * a4[1]; s[2] += xv * a4[2]; s[3] += xv * a4[3];
        s[4] += xv * b4[0]; s[5] += xv * b4[1]; s[6] += xv * b4[2]; s[7] += xv * b4[3];
    }
    #pragma unroll
    for (int r = 0; r < 8; ++r) red[t * 8 + r] = s[r];
    __syncthreads();
    for (int off = 128; off > 0; off >>= 1) {
        if (t < off)
            for (int r = 0; r < 8; ++r) red[t * 8 + r] += red[(t + off) * 8 + r];
        __syncthreads();
    }
    if (t < 4)               alpha[b * RANK + t]      = fmaxf(red[t], 0.f) + EPSV;
    else if (t < 8)          beta [b * RANK + (t-4)]  = fmaxf(red[t], 0.f) + EPSV;
}

// ---------------------------------------------------------------- gathered sel matrices
// one block per (b,h): 16 columns (i,j) x 16 k-strided partial sums, LDS reduce
__global__ void sel_kernel(const float* __restrict__ x, const float* __restrict__ w,
                           const int* __restrict__ indices, float* __restrict__ sel) {
    __shared__ float red[16][16];
    const int bh = blockIdx.x;
    const int b  = bh / NH;
    const int v  = indices[bh];
    const int t  = threadIdx.x;           // 256
    const int cid = t & 15;               // column id = i*4+j
    const int kg  = t >> 4;               // k group 0..15
    const int i = cid >> 2, j = cid & 3;
    const size_t col = (size_t)i * (VOCAB * RANK) + (size_t)v * RANK + j;
    float s = 0.f;
    for (int k = kg; k < EMB; k += 16)
        s += x[(size_t)b * EMB + k] * w[(size_t)k * NCOLS + col];
    red[kg][cid] = s;
    __syncthreads();
    if (t < 16) {
        float acc = 0.f;
        #pragma unroll
        for (int r = 0; r < 16; ++r) acc += red[r][t];
        sel[(size_t)bh * 16 + t] = fmaxf(acc, 0.f) + EPSV;
    }
}

// ---------------------------------------------------------------- finalize: per-batch nll
__global__ void finalize_kernel(const float* __restrict__ alpha, const float* __restrict__ beta,
                                const float* __restrict__ sel,   const float* __restrict__ cm_raw,
                                float* __restrict__ out) {
    const int b = blockIdx.x * blockDim.x + threadIdx.x;
    if (b >= BATCH) return;
    const float* s = sel + (size_t)b * NH * 16;
    float m[16], tmp[16], c[16], z[16];
    #pragma unroll
    for (int e = 0; e < 16; ++e) m[e] = s[e];
    for (int h = 1; h < NH; ++h) {
        const float* sh = s + h * 16;
        for (int i = 0; i < 4; ++i)
            for (int jj = 0; jj < 4; ++jj) {
                float a = 0.f;
                for (int k = 0; k < 4; ++k) a += m[i*4+k] * sh[k*4+jj];
                tmp[i*4+jj] = a;
            }
        for (int e = 0; e < 16; ++e) m[e] = tmp[e];
    }
    const float* al = alpha + b * RANK;
    const float* be = beta  + b * RANK;
    float p = 0.f;
    for (int i = 0; i < 4; ++i)
        for (int jj = 0; jj < 4; ++jj) p += al[i] * m[i*4+jj] * be[jj];

    #pragma unroll
    for (int e = 0; e < 16; ++e) { c[e] = cm_raw[b*16 + e] + (float)VOCAB * EPSV; z[e] = c[e]; }
    for (int h = 1; h < NH; ++h) {
        for (int i = 0; i < 4; ++i)
            for (int jj = 0; jj < 4; ++jj) {
                float a = 0.f;
                for (int k = 0; k < 4; ++k) a += z[i*4+k] * c[k*4+jj];
                tmp[i*4+jj] = a;
            }
        for (int e = 0; e < 16; ++e) z[e] = tmp[e];
    }
    float nc = 0.f;
    for (int i = 0; i < 4; ++i)
        for (int jj = 0; jj < 4; ++jj) nc += al[i] * z[i*4+jj] * be[jj];

    out[b] = logf(nc) - logf(p);
}

// ---------------------------------------------------------------- launch
extern "C" void kernel_launch(void* const* d_in, const int* in_sizes, int n_in,
                              void* d_out, int out_size, void* d_ws, size_t ws_size,
                              hipStream_t stream) {
    const float* x  = (const float*)d_in[0];
    const float* wa = (const float*)d_in[1];
    const float* wb = (const float*)d_in[2];
    const float* wv = (const float*)d_in[3];
    const int*  idx = (const int*)d_in[4];
    float* out = (float*)d_out;

    char* ws = (char*)d_ws;
    _Float16* xh   = (_Float16*)ws;                                 // 1 MB
    float*    cm   = (float*)(ws + (1 << 20));                      // 16 KB
    float*    alp  = (float*)(ws + (1 << 20) + 16384);              // 4 KB
    float*    bet  = alp + BATCH * RANK;                            // 4 KB
    float*    sel  = bet + BATCH * RANK;                            // 64 KB

    convert_x_kernel<<<(BATCH * EMB + 255) / 256, 256, 0, stream>>>(x, xh);
    zero_cm_kernel<<<16, 256, 0, stream>>>(cm);
    gemm_cm_kernel<<<NCOLS / NT, 512, 0, stream>>>(xh, wv, cm);
    alphabeta_kernel<<<BATCH, 256, 0, stream>>>(x, wa, wb, alp, bet);
    sel_kernel<<<BATCH * NH, 256, 0, stream>>>(x, wv, idx, sel);
    finalize_kernel<<<1, 256, 0, stream>>>(alp, bet, sel, cm, out);
}
